// GAT_12352325943365
// MI455X (gfx1250) — matrix operand order repaired and checked by reading, not compiled, not run on hardware
//
#include <hip/hip_runtime.h>
#include <hip/hip_bf16.h>

typedef __attribute__((ext_vector_type(16))) _Float16 v16h;
typedef __attribute__((ext_vector_type(8)))  _Float16 v8h;
typedef __attribute__((ext_vector_type(8)))  float    v8f;

#define NEG_SLOPE 0.2f

// ---------- helpers: monotonic float<->uint order map (for atomicMax) ----------
__device__ __forceinline__ unsigned f2ord(float f) {
  unsigned u = __float_as_uint(f);
  return (u & 0x80000000u) ? ~u : (u | 0x80000000u);
}
__device__ __forceinline__ float ord2f(unsigned u) {
  return (u & 0x80000000u) ? __uint_as_float(u & 0x7fffffffu) : __uint_as_float(~u);
}

// ---------- generic fills / conversions ----------
__global__ void fill_u32(unsigned* __restrict__ p, unsigned v, long long n) {
  long long i = (long long)blockIdx.x * blockDim.x + threadIdx.x;
  if (i < n) p[i] = v;
}

__global__ void to_f16(const float* __restrict__ x, _Float16* __restrict__ xh, long long n) {
  long long i = (long long)blockIdx.x * blockDim.x + threadIdx.x;
  if (i < n) xh[i] = (_Float16)x[i];
}

// Pack W[K, Nact] (row-major f32) into WMMA B-fragment layout:
// frag[(ks*ntiles+nt)*32 + lane][e] = W[ks*32 + (lane>>4)*16 + e, nt*16 + (lane&15)]
// zero-padded for n >= Nact.
__global__ void pack_w(const float* __restrict__ W, _Float16* __restrict__ Bp,
                       int K, int Nact, int ntiles, int ksteps) {
  int idx = blockIdx.x * blockDim.x + threadIdx.x;
  int total = ksteps * ntiles * 32 * 16;
  if (idx >= total) return;
  int e    = idx & 15;
  int lane = (idx >> 4) & 31;
  int rem  = idx >> 9;
  int nt   = rem % ntiles;
  int ks   = rem / ntiles;
  int n = nt * 16 + (lane & 15);
  int k = ks * 32 + (lane >> 4) * 16 + e;
  float v = (n < Nact && k < K) ? W[(size_t)k * Nact + n] : 0.0f;
  Bp[idx] = (_Float16)v;
}

// ---------- WMMA GEMM: C[M, NTILES*16] = A[M, KSTEPS*32] @ Bpacked ----------
template <int KSTEPS, int NTILES>
__global__ __launch_bounds__(256) void gemm_wmma(const _Float16* __restrict__ A,
                                                 const _Float16* __restrict__ Bp,
                                                 float* __restrict__ C, int Mtiles) {
  const int gwave = (int)(((long long)blockIdx.x * blockDim.x + threadIdx.x) >> 5);
  const int lane  = threadIdx.x & 31;
  if (gwave >= Mtiles) return;
  const int rlo   = lane & 15;   // row-in-tile (A) / column-in-tile (B,C)
  const int khalf = lane >> 4;   // lane half selects K sub-range / M half of C
  const int K = KSTEPS * 32;
  const long long row = (long long)gwave * 16 + rlo;

  v8f z = {};
  v8f acc[NTILES];
#pragma unroll
  for (int i = 0; i < NTILES; ++i) acc[i] = z;

  const v16h* __restrict__ bfrag = (const v16h*)Bp;
#pragma unroll
  for (int ks = 0; ks < KSTEPS; ++ks) {
    // A fragment (contiguous K order, matching packed B): lanes 0-15 hold K=[0..15];
    // lanes 16-31 hold K=[16..31] of this 32-wide K step.
    const _Float16* ap = A + row * K + ks * 32 + khalf * 16;
    v8h lo = *(const v8h*)(ap);
    v8h hi = *(const v8h*)(ap + 8);
    v16h a;
#pragma unroll
    for (int i = 0; i < 8; ++i) { a[i] = lo[i]; a[i + 8] = hi[i]; }
#pragma unroll
    for (int nt = 0; nt < NTILES; ++nt) {
      v16h b = bfrag[(ks * NTILES + nt) * 32 + lane];
      acc[nt] = __builtin_amdgcn_wmma_f32_16x16x32_f16(
          false, a, false, b, (short)0, acc[nt], false, false);
    }
  }

  // C/D layout: lanes 0-15 -> M = vgpr, lanes 16-31 -> M = vgpr+8; N = lane&15
  const int ldc = NTILES * 16;
  float* crow = C + (long long)gwave * 16 * ldc;
#pragma unroll
  for (int nt = 0; nt < NTILES; ++nt) {
#pragma unroll
    for (int v = 0; v < 8; ++v) {
      int m = khalf * 8 + v;
      crow[(long long)m * ldc + nt * 16 + rlo] = acc[nt][v];
    }
  }
}

// ---------- per-row dot products: alpha_src[r] = H[r,:] . a_src ----------
__global__ void row_dots(const float* __restrict__ H, const float* __restrict__ av,
                         const float* __restrict__ dv, float* __restrict__ outs,
                         float* __restrict__ outd, int rows, int C, int ldH) {
  int gwave = (int)(((long long)blockIdx.x * blockDim.x + threadIdx.x) >> 5);
  int lane = threadIdx.x & 31;
  if (gwave >= rows) return;
  const float* h = H + (long long)gwave * ldH;
  float ss = 0.f, dd = 0.f;
  for (int c = lane; c < C; c += 32) {
    float v = h[c];
    ss += v * av[c];
    dd += v * dv[c];
  }
  for (int off = 16; off; off >>= 1) {
    ss += __shfl_down(ss, off, 32);
    dd += __shfl_down(dd, off, 32);
  }
  if (lane == 0) { outs[gwave] = ss; outd[gwave] = dd; }
}

// ---------- edge passes ----------
__device__ __forceinline__ void edge_ids(const long long* __restrict__ ei, long long E,
                                         long long eidx, long long& s, long long& d) {
  if (eidx < E) { s = ei[eidx]; d = ei[E + eidx]; }
  else          { s = eidx - E; d = s; }           // self-loops appended
}

__global__ void edge_pre(const long long* __restrict__ ei, long long E, long long Nn,
                         const float* __restrict__ as, const float* __restrict__ ad,
                         float* __restrict__ eraw, unsigned* __restrict__ mu) {
  long long idx = (long long)blockIdx.x * blockDim.x + threadIdx.x;
  if (idx >= E + Nn) return;
  long long s, d;
  edge_ids(ei, E, idx, s, d);
  float e = as[s] + ad[d];
  e = (e > 0.f) ? e : NEG_SLOPE * e;     // leaky_relu
  eraw[idx] = e;
  atomicMax(&mu[d], f2ord(e));           // segment max (ordered-uint)
}

__global__ void edge_expsum(const long long* __restrict__ ei, long long E, long long Nn,
                            float* __restrict__ w, const unsigned* __restrict__ mu,
                            float* __restrict__ sb) {
  long long idx = (long long)blockIdx.x * blockDim.x + threadIdx.x;
  if (idx >= E + Nn) return;
  long long s, d;
  edge_ids(ei, E, idx, s, d);
  float ww = __expf(w[idx] - ord2f(mu[d]));
  w[idx] = ww;
  atomicAdd(&sb[d], ww);
}

__global__ void edge_norm(const long long* __restrict__ ei, long long E, long long Nn,
                          float* __restrict__ w, const float* __restrict__ sb) {
  long long idx = (long long)blockIdx.x * blockDim.x + threadIdx.x;
  if (idx >= E + Nn) return;
  long long s, d;
  edge_ids(ei, E, idx, s, d);
  w[idx] = w[idx] / sb[d];
}

template <int C>
__global__ void edge_agg(const long long* __restrict__ ei, long long E, long long Nn,
                         const float* __restrict__ alpha, const float* __restrict__ H,
                         int ldH, float* __restrict__ out) {
  long long idx = (long long)blockIdx.x * blockDim.x + threadIdx.x;
  long long tot = (E + Nn) * C;
  if (idx >= tot) return;
  long long eidx = idx / C;
  int c = (int)(idx - eidx * (long long)C);
  long long s, d;
  edge_ids(ei, E, eidx, s, d);
  atomicAdd(&out[d * (long long)C + c], alpha[eidx] * H[s * (long long)ldH + c]);
}

// ---------- bias + relu + convert to f16 (layer-2 input) ----------
__global__ void bias_relu_h16(const float* __restrict__ agg, const float* __restrict__ b,
                              _Float16* __restrict__ xh, long long total, int C) {
  long long i = (long long)blockIdx.x * blockDim.x + threadIdx.x;
  if (i >= total) return;
  int c = (int)(i % C);
  float v = agg[i] + b[c];
  xh[i] = (_Float16)(v > 0.f ? v : 0.f);
}

// ---------- final: out = log_softmax(agg + b) over 40 channels ----------
__global__ void final_lsm(const float* __restrict__ agg, const float* __restrict__ b,
                          float* __restrict__ out, int rows) {
  int gwave = (int)(((long long)blockIdx.x * blockDim.x + threadIdx.x) >> 5);
  int lane = threadIdx.x & 31;
  if (gwave >= rows) return;
  const float* r = agg + (long long)gwave * 40;
  bool hi = lane < 8;
  float x0 = r[lane] + b[lane];
  float x1 = hi ? r[lane + 32] + b[lane + 32] : -3.4e38f;
  float mx = fmaxf(x0, x1);
  for (int off = 16; off; off >>= 1) mx = fmaxf(mx, __shfl_xor(mx, off, 32));
  float se = __expf(x0 - mx) + (hi ? __expf(x1 - mx) : 0.f);
  for (int off = 16; off; off >>= 1) se += __shfl_xor(se, off, 32);
  float lse = mx + logf(se);
  out[(long long)gwave * 40 + lane] = x0 - lse;
  if (hi) out[(long long)gwave * 40 + lane + 32] = x1 - lse;
}

// ------------------------------------------------------------------
extern "C" void kernel_launch(void* const* d_in, const int* in_sizes, int n_in,
                              void* d_out, int out_size, void* d_ws, size_t ws_size,
                              hipStream_t stream) {
  (void)in_sizes; (void)n_in; (void)out_size; (void)ws_size;
  const float*      x   = (const float*)d_in[0];
  const long long*  ei  = (const long long*)d_in[1];   // [2, E] int64
  const float*      W1  = (const float*)d_in[3];
  const float*      aS1 = (const float*)d_in[4];
  const float*      aD1 = (const float*)d_in[5];
  const float*      b1  = (const float*)d_in[6];
  const float*      W2  = (const float*)d_in[7];
  const float*      aS2 = (const float*)d_in[8];
  const float*      aD2 = (const float*)d_in[9];
  const float*      b2  = (const float*)d_in[10];
  float*            out = (float*)d_out;

  const long long N = 50000, E = 800000, EP = E + N;
  const int INC = 128, HID = 96, OUTC = 40, OUTP = 48;  // OUTP: padded to 3 N-tiles
  const int MT = (int)(N / 16);                          // 3125 row tiles (exact)

  char* ws = (char*)d_ws;
  size_t off = 0;
  auto alloc = [&](size_t bytes) -> void* {
    void* p = ws + off;
    off = (off + bytes + 255) & ~(size_t)255;
    return p;
  };
  _Float16* x1h  = (_Float16*)alloc((size_t)N * INC * 2);
  _Float16* w1p  = (_Float16*)alloc((size_t)4 * 6 * 32 * 16 * 2);
  _Float16* w2p  = (_Float16*)alloc((size_t)3 * 3 * 32 * 16 * 2);
  float*    h1   = (float*)alloc((size_t)N * HID * 4);
  float*    h2   = (float*)alloc((size_t)N * OUTP * 4);
  float*    asb  = (float*)alloc((size_t)N * 4);
  float*    adb  = (float*)alloc((size_t)N * 4);
  unsigned* mu   = (unsigned*)alloc((size_t)N * 4);
  float*    sb   = (float*)alloc((size_t)N * 4);
  float*    wed  = (float*)alloc((size_t)EP * 4);
  float*    agg1 = (float*)alloc((size_t)N * HID * 4);
  _Float16* x2h  = (_Float16*)alloc((size_t)N * HID * 2);
  float*    agg2 = (float*)alloc((size_t)N * OUTC * 4);

  const int TB = 256;
  auto gs = [&](long long n) { return (unsigned)((n + TB - 1) / TB); };

  // ---- shared prep ----
  to_f16<<<gs(N * INC), TB, 0, stream>>>(x, x1h, N * INC);
  pack_w<<<gs(4 * 6 * 512), TB, 0, stream>>>(W1, w1p, INC, HID, 6, 4);
  pack_w<<<gs(3 * 3 * 512), TB, 0, stream>>>(W2, w2p, HID, OUTC, 3, 3);

  // ---- layer 1 ----
  gemm_wmma<4, 6><<<gs((long long)MT * 32), TB, 0, stream>>>(x1h, w1p, h1, MT);
  row_dots<<<gs(N * 32), TB, 0, stream>>>(h1, aS1, aD1, asb, adb, (int)N, HID, HID);
  fill_u32<<<gs(N), TB, 0, stream>>>(mu, 0u, N);
  fill_u32<<<gs(N), TB, 0, stream>>>((unsigned*)sb, 0u, N);
  fill_u32<<<gs(N * HID), TB, 0, stream>>>((unsigned*)agg1, 0u, N * HID);
  edge_pre   <<<gs(EP), TB, 0, stream>>>(ei, E, N, asb, adb, wed, mu);
  edge_expsum<<<gs(EP), TB, 0, stream>>>(ei, E, N, wed, mu, sb);
  edge_norm  <<<gs(EP), TB, 0, stream>>>(ei, E, N, wed, sb);
  edge_agg<96><<<gs(EP * 96), TB, 0, stream>>>(ei, E, N, wed, h1, HID, agg1);
  bias_relu_h16<<<gs(N * HID), TB, 0, stream>>>(agg1, b1, x2h, N * HID, HID);

  // ---- layer 2 ----
  gemm_wmma<3, 3><<<gs((long long)MT * 32), TB, 0, stream>>>(x2h, w2p, h2, MT);
  row_dots<<<gs(N * 32), TB, 0, stream>>>(h2, aS2, aD2, asb, adb, (int)N, OUTC, OUTP);
  fill_u32<<<gs(N), TB, 0, stream>>>(mu, 0u, N);
  fill_u32<<<gs(N), TB, 0, stream>>>((unsigned*)sb, 0u, N);
  fill_u32<<<gs(N * OUTC), TB, 0, stream>>>((unsigned*)agg2, 0u, N * OUTC);
  edge_pre   <<<gs(EP), TB, 0, stream>>>(ei, E, N, asb, adb, wed, mu);
  edge_expsum<<<gs(EP), TB, 0, stream>>>(ei, E, N, wed, mu, sb);
  edge_norm  <<<gs(EP), TB, 0, stream>>>(ei, E, N, wed, sb);
  edge_agg<40><<<gs(EP * 40), TB, 0, stream>>>(ei, E, N, wed, h2, OUTP, agg2);

  // ---- log_softmax ----
  final_lsm<<<gs(N * 32), TB, 0, stream>>>(agg2, b2, out, (int)N);
}
